// RNN_60241211294103
// MI455X (gfx1250) — matrix operand order repaired and checked by reading, not compiled, not run on hardware
//
#include <hip/hip_runtime.h>
#include <math.h>

// ---------------------------------------------------------------------------
// PredRNN forward on MI455X (gfx1250): implicit-GEMM conv via v_wmma bf16,
// with TDM (tensor_load_to_lds) staging of input rows into LDS.
// ---------------------------------------------------------------------------

typedef __attribute__((ext_vector_type(16))) __bf16 v16bf;
typedef __attribute__((ext_vector_type(8)))  float  v8f;
typedef __attribute__((ext_vector_type(4)))  unsigned int v4u;
typedef __attribute__((ext_vector_type(8)))  int v8i;
typedef __attribute__((ext_vector_type(4)))  int v4i;

static constexpr int B_  = 8;
static constexpr int T_  = 12;
static constexpr int H_  = 64;
static constexpr int W_  = 64;
static constexpr int C_  = 16;
static constexpr int NH_ = 64;
static constexpr int L_  = 4;
static constexpr int HP_ = H_ + 4;   // 68 (padded rows)
static constexpr int WP_ = W_ + 4;   // 68 (padded cols)
static constexpr int HW_ = H_ * W_;  // 4096
static constexpr int GCH = 7 * NH_;  // 448 gate channels
static constexpr int TSTEPS = T_ - 1;          // 11
static constexpr long long OUTN = (long long)B_ * TSTEPS * HW_ * C_;

__device__ __forceinline__ unsigned short f2bf(float f) {
  unsigned int u = __float_as_uint(f);
  unsigned int r = (u + 0x7FFFu + ((u >> 16) & 1u)) >> 16;   // RNE
  return (unsigned short)r;
}
__device__ __forceinline__ float sigmoidf_(float x) {
  return 1.0f / (1.0f + expf(-x));
}

// ---------------------------------------------------------------------------
// Weight packing: fp32 OIHW -> bf16 16x32 A-fragments (contiguous k order).
// Fragment order: [co_tile][tap][ci_chunk], each 32 lanes x 8 dwords (1KB).
// A k order used here (contiguous, not AMD's table order): lane&15 = row M; lanes 0-15 hold
// K 0..15 in dwords 0..7; lanes 16-31 hold K 16..31 (matches B fill).
// ---------------------------------------------------------------------------
__global__ void pack_weights(const float* __restrict__ w, unsigned int* __restrict__ out,
                             int Cout, int Cin, int taps, int nfrags) {
  int tid = blockIdx.x * blockDim.x + threadIdx.x;
  if (tid >= nfrags * 256) return;
  int frag = tid >> 8, r = tid & 255;
  int lane = r >> 3, d = r & 7;
  int chunks = (Cin + 31) >> 5;
  int ct  = frag / (taps * chunks);
  int rem = frag % (taps * chunks);
  int tap = rem / chunks, cc = rem % chunks;
  int co = ct * 16 + (lane & 15);
  int kb = (lane < 16) ? 2 * d : 16 + 2 * d;
  int ci = cc * 32 + kb;
  unsigned short lo = 0, hi = 0;
  if (co < Cout) {
    if (ci < Cin)     lo = f2bf(w[(long long)(co * Cin + ci)     * taps + tap]);
    if (ci + 1 < Cin) hi = f2bf(w[(long long)(co * Cin + ci + 1) * taps + tap]);
  }
  out[(long long)frag * 256 + lane * 8 + d] = (unsigned)lo | ((unsigned)hi << 16);
}

// ---------------------------------------------------------------------------
// WMMA implicit-GEMM conv. Input = zero-padded HWC bf16 (HP x WP x cpitch).
// Grid: (y=64, co_tile_group, batch). Each wave: 16 Cout x 64 px (one row).
// STAGE=true: one TDM descriptor pulls the 5 contiguous padded input rows
// into LDS (async, TENSORcnt), shared by the block's 4 Cout-tile waves;
// B fragments then come from ds_load_b128.
// MODE 0: store  fp32 NCHW gates      MODE 1: accumulate fp32 NCHW gates
// MODE 2: h = sigmoid(o_gate)*tanh(acc) -> padded HWC bf16 (wcl 1x1 conv)
// MODE 3: x_gen -> fp32 HWC scratch + d_out frames (w_last 1x1 conv)
// ---------------------------------------------------------------------------
struct ConvParams {
  const unsigned int*   wpack;
  const unsigned short* in;        // padded HWC bf16
  long long in_bstride;            // elements per batch image
  int cpitch, ci_off, Cin, taps;
  float* gates;                    // fp32 NCHW [B][448][HW]
  int co_base, co_split, co_add;
  unsigned short* h_out;           // MODE 2
  float* xgen;                     // MODE 3
  float* frames_out;               // MODE 3 (d_out base)
  int t;                           // MODE 3
};

template <int MODE, bool STAGE>
__global__ __launch_bounds__(128) void conv_wmma(ConvParams p) {
#if defined(__gfx1250__)
  const int lane = threadIdx.x & 31;
  const int wv   = threadIdx.x >> 5;
  const int y    = blockIdx.x;
  const int b    = blockIdx.z;
  const int ct   = blockIdx.y * (blockDim.x >> 5) + wv;
  const int pcol = lane & 15;
  const int hl   = lane >> 4;

  extern __shared__ unsigned short smem[];
  if (STAGE) {
    if (wv == 0) {
      // 1-D TDM: rows y..y+4 of padded image are contiguous in memory.
      const unsigned long long ga = (unsigned long long)(const void*)(
          p.in + (long long)b * p.in_bstride + (long long)y * WP_ * p.cpitch);
      const unsigned n8 = (unsigned)((5u * WP_ * p.cpitch * 2u) >> 3);  // 8B units
      v4u g0;
      g0[0] = 1u;                                   // count=1 (valid descriptor)
      g0[1] = (unsigned)(size_t)smem;               // lds_addr (bytes)
      g0[2] = (unsigned)(ga & 0xffffffffull);       // global_addr[31:0]
      g0[3] = (unsigned)((ga >> 32) & 0x01ffffffull) | 0x80000000u;  // addr[56:32]|type=2
      v8i g1 = {};
      g1[0] = (int)(3u << 16);                      // data_size = 8B
      g1[1] = (int)((n8 & 0xffffu) << 16);          // tensor_dim0[15:0]
      g1[2] = (int)(((n8 >> 16) & 0xffffu) | (1u << 16));  // td0[31:16], tensor_dim1=1
      g1[3] = (int)((n8 & 0xffffu) << 16);          // tile_dim0 = n8 (<= 10880)
      g1[4] = 0;                                    // tile_dim1/2 unused
      g1[5] = (int)n8;                              // tensor_dim0_stride lo
      g1[6] = 0;
      g1[7] = 0;
      v4i gz = {};
#if defined(__clang_major__) && (__clang_major__ >= 23)
      v8i gz8 = {};
      __builtin_amdgcn_tensor_load_to_lds(g0, g1, gz, gz, gz8, 0);
#else
      __builtin_amdgcn_tensor_load_to_lds(g0, g1, gz, gz, 0);
#endif
      __builtin_amdgcn_s_wait_tensorcnt(0);
    }
    __syncthreads();
  }

  v8f acc[4] = {v8f{}, v8f{}, v8f{}, v8f{}};
  const int chunks = (p.Cin + 31) >> 5;
  const unsigned int* wfrag =
      p.wpack + ((long long)ct * p.taps * chunks) * 256 + lane * 8;
  const unsigned short* inb =
      p.in + (long long)b * p.in_bstride + p.ci_off + hl * 16;

  for (int tap = 0; tap < p.taps; ++tap) {
    const int kh = (p.taps == 25) ? tap / 5 : 2;
    const int kw = (p.taps == 25) ? tap % 5 : 2;
    const unsigned short* inrow = inb + (long long)((y + kh) * WP_ + kw) * p.cpitch;
    const int lrow = (kh * WP_ + kw) * p.cpitch + p.ci_off + hl * 16;
    for (int cc = 0; cc < chunks; ++cc) {
      union { uint4 q[2]; v16bf v; } A;
      A.q[0] = *(const uint4*)(wfrag);
      A.q[1] = *(const uint4*)(wfrag + 4);
      __builtin_prefetch(wfrag + 256, 0, 1);
      wfrag += 256;
      const bool bvalid = (cc * 32 + hl * 16) < p.Cin;
      const int coff = cc * 32;
#pragma unroll
      for (int s = 0; s < 4; ++s) {
        union { uint4 q[2]; v16bf v; } Bf;
        if (bvalid) {
          // B k order used here (32x16 bf16, same as A): lane&15 = column N (pixel);
          // lanes 0-15 hold K=0..15, lanes 16-31 hold K=16..31 (channel-contig).
          if (STAGE) {
            const unsigned short* src = smem + lrow + (s * 16 + pcol) * p.cpitch + coff;
            Bf.q[0] = *(const uint4*)(src);
            Bf.q[1] = *(const uint4*)(src + 8);
          } else {
            const unsigned short* src = inrow + (long long)(s * 16 + pcol) * p.cpitch + coff;
            Bf.q[0] = *(const uint4*)(src);
            Bf.q[1] = *(const uint4*)(src + 8);
          }
        } else {
          Bf.q[0] = make_uint4(0, 0, 0, 0);
          Bf.q[1] = make_uint4(0, 0, 0, 0);
        }
        acc[s] = __builtin_amdgcn_wmma_f32_16x16x32_bf16(
            false, A.v, false, Bf.v, (short)0, acc[s], false, false);
      }
    }
  }

  // D layout: lane&15 = N (pixel), vgpr v -> M = v + 8*(lane>>4).
  const int px0 = y * W_;
  if (MODE == 0 || MODE == 1) {
    float* gb = p.gates + (long long)b * GCH * HW_;
#pragma unroll
    for (int s = 0; s < 4; ++s) {
      const int px = px0 + s * 16 + pcol;
#pragma unroll
      for (int v = 0; v < 8; ++v) {
        const int co = ct * 16 + v + 8 * hl;
        const int oc = p.co_base + co + (co >= p.co_split ? p.co_add : 0);
        float* dst = gb + (long long)oc * HW_ + px;
        if (MODE == 1) *dst += acc[s][v]; else *dst = acc[s][v];
      }
    }
  } else if (MODE == 2) {
    const float* og = p.gates + (long long)b * GCH * HW_ + (long long)384 * HW_;
#pragma unroll
    for (int s = 0; s < 4; ++s) {
      const int x  = s * 16 + pcol;
      const int px = px0 + x;
      unsigned short hs[8];
#pragma unroll
      for (int v = 0; v < 8; ++v) {
        const int co = ct * 16 + v + 8 * hl;
        const float o = og[(long long)co * HW_ + px];
        hs[v] = f2bf(sigmoidf_(o) * tanhf(acc[s][v]));
      }
      unsigned short* dst = p.h_out + (long long)b * (HP_ * WP_ * NH_) +
                            (long long)((y + 2) * WP_ + x + 2) * NH_ + ct * 16 + 8 * hl;
      *(uint4*)dst = *(const uint4*)hs;
    }
  } else {  // MODE 3: Cout=16, ct==0, channel c = v + 8*hl
#pragma unroll
    for (int s = 0; s < 4; ++s) {
      const int x = s * 16 + pcol;
      float4 f0 = make_float4(acc[s][0], acc[s][1], acc[s][2], acc[s][3]);
      float4 f1 = make_float4(acc[s][4], acc[s][5], acc[s][6], acc[s][7]);
      float* xd = p.xgen + ((long long)b * HW_ + px0 + x) * C_ + 8 * hl;
      *(float4*)xd = f0;
      *((float4*)xd + 1) = f1;
      float* od = p.frames_out +
                  (((long long)(b * TSTEPS + p.t)) * HW_ + px0 + x) * C_ + 8 * hl;
      *(float4*)od = f0;
      *((float4*)od + 1) = f1;
    }
  }
#else
  (void)p;
#endif
}

// ---------------------------------------------------------------------------
// Elementwise ST-LSTM gates: c_new, m_new; mem -> padded HWC bf16 (128 ch).
// ---------------------------------------------------------------------------
__global__ void gates_kernel(const float* __restrict__ gates, float* __restrict__ cbuf,
                             float* __restrict__ mbuf, unsigned short* __restrict__ mem_hwc) {
  int idx = blockIdx.x * blockDim.x + threadIdx.x;
  if (idx >= B_ * NH_ * HW_) return;
  const int px = idx & (HW_ - 1);
  const int ch = (idx >> 12) & 63;
  const int b  = idx >> 18;
  const float* gb = gates + (long long)b * GCH * HW_ + px;
  const float gi  = gb[(long long)(ch)       * HW_];
  const float gf  = gb[(long long)(64 + ch)  * HW_];
  const float gg  = gb[(long long)(128 + ch) * HW_];
  const float gip = gb[(long long)(192 + ch) * HW_];
  const float gfp = gb[(long long)(256 + ch) * HW_];
  const float ggp = gb[(long long)(320 + ch) * HW_];
  const long long ci = (long long)b * NH_ * HW_ + (long long)ch * HW_ + px;
  const float c_old = cbuf[ci], m_old = mbuf[ci];
  const float c_new = sigmoidf_(gf + 1.0f) * c_old + sigmoidf_(gi) * tanhf(gg);
  const float m_new = sigmoidf_(gfp + 1.0f) * m_old + sigmoidf_(gip) * tanhf(ggp);
  cbuf[ci] = c_new;
  mbuf[ci] = m_new;
  const int y = px >> 6, x = px & 63;
  unsigned short* mm = mem_hwc + (long long)b * (HP_ * WP_ * 2 * NH_) +
                       (long long)((y + 2) * WP_ + x + 2) * (2 * NH_);
  mm[ch]       = f2bf(c_new);
  mm[64 + ch]  = f2bf(m_new);
}

// ---------------------------------------------------------------------------
// Scheduled-sampling blend -> padded HWC bf16 (16 ch).
// ---------------------------------------------------------------------------
__global__ void blend_kernel(const float* __restrict__ frames, const float* __restrict__ mask,
                             const float* __restrict__ xgen, unsigned short* __restrict__ net_hwc,
                             int t) {
  int idx = blockIdx.x * blockDim.x + threadIdx.x;
  if (idx >= B_ * HW_ * C_) return;
  const int c  = idx & 15;
  const int px = (idx >> 4) & (HW_ - 1);
  const int b  = idx >> 16;
  const float fr = frames[(((long long)b * T_ + t) * HW_ + px) * C_ + c];
  float mv = 1.0f;
  if (t >= 6) mv = mask[((long long)b * 5 + (t - 6)) * HW_ + px];
  const float xg  = xgen[((long long)b * HW_ + px) * C_ + c];
  const float net = mv * fr + (1.0f - mv) * xg;
  const int y = px >> 6, x = px & 63;
  net_hwc[(long long)b * (HP_ * WP_ * C_) + (long long)((y + 2) * WP_ + x + 2) * C_ + c] =
      f2bf(net);
}

__global__ void zero_kernel(unsigned int* __restrict__ p, long long n) {
  long long i = (long long)blockIdx.x * blockDim.x + threadIdx.x;
  const long long stride = (long long)gridDim.x * blockDim.x;
  for (; i < n; i += stride) p[i] = 0u;
}

// ---------------------------------------------------------------------------
// MSE loss (two-stage reduction with LDS).
// ---------------------------------------------------------------------------
__global__ void loss_partial(const float* __restrict__ out, const float* __restrict__ frames,
                             float* __restrict__ partial) {
  __shared__ float sdata[256];
  float s = 0.0f;
  for (long long i = (long long)blockIdx.x * blockDim.x + threadIdx.x; i < OUTN;
       i += (long long)gridDim.x * blockDim.x) {
    const int c  = (int)(i & 15);
    const long long r = i >> 4;
    const int px = (int)(r & (HW_ - 1));
    const long long r2 = r >> 12;
    const int t = (int)(r2 % TSTEPS);
    const int b = (int)(r2 / TSTEPS);
    const float f = frames[(((long long)b * T_ + t + 1) * HW_ + px) * C_ + c];
    const float d = out[i] - f;
    s += d * d;
  }
  sdata[threadIdx.x] = s;
  __syncthreads();
  for (int k = 128; k > 0; k >>= 1) {
    if ((int)threadIdx.x < k) sdata[threadIdx.x] += sdata[threadIdx.x + k];
    __syncthreads();
  }
  if (threadIdx.x == 0) partial[blockIdx.x] = sdata[0];
}

__global__ void loss_final(const float* __restrict__ partial, float* __restrict__ out_loss,
                           int nb) {
  __shared__ float sdata[256];
  float s = 0.0f;
  for (int i = threadIdx.x; i < nb; i += 256) s += partial[i];
  sdata[threadIdx.x] = s;
  __syncthreads();
  for (int k = 128; k > 0; k >>= 1) {
    if ((int)threadIdx.x < k) sdata[threadIdx.x] += sdata[threadIdx.x + k];
    __syncthreads();
  }
  if (threadIdx.x == 0) *out_loss = sdata[0] / (float)OUTN;
}

// ---------------------------------------------------------------------------
// Host driver.
// ---------------------------------------------------------------------------
static inline long long nfrags(int Cout, int Cin, int taps) {
  return (long long)(Cout / 16) * taps * ((Cin + 31) / 32);
}

extern "C" void kernel_launch(void* const* d_in, const int* in_sizes, int n_in,
                              void* d_out, int out_size, void* d_ws, size_t ws_size,
                              hipStream_t stream) {
  (void)in_sizes; (void)n_in; (void)out_size; (void)ws_size;
  const float* frames  = (const float*)d_in[0];
  const float* mask    = (const float*)d_in[1];
  const float* wx0     = (const float*)d_in[2];
  const float* wx_rest = (const float*)d_in[3];
  const float* wh      = (const float*)d_in[4];
  const float* wm      = (const float*)d_in[5];
  const float* wo      = (const float*)d_in[6];
  const float* wcl     = (const float*)d_in[7];
  const float* wlast   = (const float*)d_in[8];
  float* out = (float*)d_out;

  char* ws = (char*)d_ws;
  size_t off = 0;
  auto alloc = [&](size_t bytes) -> void* {
    off = (off + 255) & ~(size_t)255;
    void* p = ws + off;
    off += bytes;
    return p;
  };

  // ---- packed weights ----
  unsigned int* pk_wx0 = (unsigned int*)alloc(nfrags(GCH, C_, 25) * 1024);
  unsigned int* pk_wx[3];
  for (int l = 0; l < 3; ++l) pk_wx[l] = (unsigned int*)alloc(nfrags(GCH, NH_, 25) * 1024);
  unsigned int* pk_wh[4];
  for (int l = 0; l < 4; ++l) pk_wh[l] = (unsigned int*)alloc(nfrags(4 * NH_, NH_, 25) * 1024);
  unsigned int* pk_wm[4];
  for (int l = 0; l < 4; ++l) pk_wm[l] = (unsigned int*)alloc(nfrags(3 * NH_, NH_, 25) * 1024);
  unsigned int* pk_wo[4];
  for (int l = 0; l < 4; ++l) pk_wo[l] = (unsigned int*)alloc(nfrags(NH_, 2 * NH_, 25) * 1024);
  unsigned int* pk_wcl[4];
  for (int l = 0; l < 4; ++l) pk_wcl[l] = (unsigned int*)alloc(nfrags(NH_, 2 * NH_, 1) * 1024);
  unsigned int* pk_wlast = (unsigned int*)alloc(nfrags(C_, NH_, 1) * 1024);

  // ---- activations / state ----
  float* gates = (float*)alloc((size_t)B_ * GCH * HW_ * 4);
  size_t state_begin = (off + 255) & ~(size_t)255;
  float* cbuf[4];
  for (int l = 0; l < 4; ++l) cbuf[l] = (float*)alloc((size_t)B_ * NH_ * HW_ * 4);
  float* mbuf = (float*)alloc((size_t)B_ * NH_ * HW_ * 4);
  unsigned short* h_hwc[4];
  for (int l = 0; l < 4; ++l) h_hwc[l] = (unsigned short*)alloc((size_t)B_ * HP_ * WP_ * NH_ * 2);
  unsigned short* mem_hwc = (unsigned short*)alloc((size_t)B_ * HP_ * WP_ * 2 * NH_ * 2);
  unsigned short* net_hwc = (unsigned short*)alloc((size_t)B_ * HP_ * WP_ * C_ * 2);
  float* xgen = (float*)alloc((size_t)B_ * HW_ * C_ * 4);
  size_t state_end = off;
  float* partial = (float*)alloc(1024 * 4);

  // ---- pack all weights (once per launch; deterministic) ----
  auto pack = [&](const float* w, unsigned int* dst, int Cout, int Cin, int taps) {
    const long long nf = nfrags(Cout, Cin, taps);
    const long long nthr = nf * 256;
    pack_weights<<<dim3((unsigned)((nthr + 255) / 256)), dim3(256), 0, stream>>>(
        w, dst, Cout, Cin, taps, (int)nf);
  };
  pack(wx0, pk_wx0, GCH, C_, 25);
  for (int l = 0; l < 3; ++l) pack(wx_rest + (long long)l * GCH * NH_ * 25, pk_wx[l], GCH, NH_, 25);
  for (int l = 0; l < 4; ++l) pack(wh + (long long)l * 4 * NH_ * NH_ * 25, pk_wh[l], 4 * NH_, NH_, 25);
  for (int l = 0; l < 4; ++l) pack(wm + (long long)l * 3 * NH_ * NH_ * 25, pk_wm[l], 3 * NH_, NH_, 25);
  for (int l = 0; l < 4; ++l) pack(wo + (long long)l * NH_ * 2 * NH_ * 25, pk_wo[l], NH_, 2 * NH_, 25);
  for (int l = 0; l < 4; ++l) pack(wcl + (long long)l * NH_ * 2 * NH_, pk_wcl[l], NH_, 2 * NH_, 1);
  pack(wlast, pk_wlast, C_, NH_, 1);

  // ---- zero all recurrent state + padded halos (one pass) ----
  {
    long long nwords = (long long)(state_end - state_begin) / 4;
    zero_kernel<<<dim3(2048), dim3(256), 0, stream>>>(
        (unsigned int*)(ws + state_begin), nwords);
  }

  const long long hstride   = (long long)HP_ * WP_ * NH_;
  const long long memstride = (long long)HP_ * WP_ * 2 * NH_;
  const long long netstride = (long long)HP_ * WP_ * C_;
  // LDS bytes for staged kernels: 5 padded rows of (WP_ x cpitch) bf16.
  auto stage_bytes = [](int cpitch) { return (size_t)(5 * WP_ * cpitch * 2); };

  // ---- recurrent scan ----
  for (int t = 0; t < TSTEPS; ++t) {
    blend_kernel<<<dim3((B_ * HW_ * C_ + 255) / 256), dim3(256), 0, stream>>>(
        frames, mask, xgen, net_hwc, t);

    for (int l = 0; l < L_; ++l) {
      // xcat: write all 448 gate channels
      ConvParams cx{};
      cx.gates = gates; cx.taps = 25;
      cx.co_base = 0; cx.co_split = 1 << 30; cx.co_add = 0; cx.ci_off = 0;
      if (l == 0) {
        cx.wpack = pk_wx0; cx.in = net_hwc; cx.cpitch = C_; cx.Cin = C_;
        cx.in_bstride = netstride;
      } else {
        cx.wpack = pk_wx[l - 1]; cx.in = h_hwc[l - 1]; cx.cpitch = NH_; cx.Cin = NH_;
        cx.in_bstride = hstride;
      }
      conv_wmma<0, true><<<dim3(64, GCH / 64, B_), dim3(128), stage_bytes(cx.cpitch), stream>>>(cx);

      // hcat: accumulate into [0..191] and [384..447]
      ConvParams chh{};
      chh.gates = gates; chh.taps = 25;
      chh.wpack = pk_wh[l]; chh.in = h_hwc[l]; chh.cpitch = NH_; chh.Cin = NH_;
      chh.in_bstride = hstride; chh.ci_off = 0;
      chh.co_base = 0; chh.co_split = 192; chh.co_add = 192;
      conv_wmma<1, true><<<dim3(64, (4 * NH_) / 64, B_), dim3(128), stage_bytes(NH_), stream>>>(chh);

      // mcat: accumulate into [192..383] (reads m = mem channels 64..127)
      ConvParams cmm{};
      cmm.gates = gates; cmm.taps = 25;
      cmm.wpack = pk_wm[l]; cmm.in = mem_hwc; cmm.cpitch = 2 * NH_; cmm.Cin = NH_;
      cmm.in_bstride = memstride; cmm.ci_off = NH_;
      cmm.co_base = 192; cmm.co_split = 1 << 30; cmm.co_add = 0;
      conv_wmma<1, true><<<dim3(64, (3 * NH_) / 64, B_), dim3(128), stage_bytes(2 * NH_), stream>>>(cmm);

      // gates -> c_new, m_new, mem (bf16 HWC)
      gates_kernel<<<dim3((B_ * NH_ * HW_ + 255) / 256), dim3(256), 0, stream>>>(
          gates, cbuf[l], mbuf, mem_hwc);

      // o conv(mem): accumulate into [384..447]
      ConvParams coo{};
      coo.gates = gates; coo.taps = 25;
      coo.wpack = pk_wo[l]; coo.in = mem_hwc; coo.cpitch = 2 * NH_; coo.Cin = 2 * NH_;
      coo.in_bstride = memstride; coo.ci_off = 0;
      coo.co_base = 384; coo.co_split = 1 << 30; coo.co_add = 0;
      conv_wmma<1, true><<<dim3(64, NH_ / 64, B_), dim3(128), stage_bytes(2 * NH_), stream>>>(coo);

      // wcl 1x1 conv(mem) with fused sigmoid(o)*tanh epilogue -> h_hwc[l]
      ConvParams ccl{};
      ccl.gates = gates; ccl.taps = 1;
      ccl.wpack = pk_wcl[l]; ccl.in = mem_hwc; ccl.cpitch = 2 * NH_; ccl.Cin = 2 * NH_;
      ccl.in_bstride = memstride; ccl.ci_off = 0;
      ccl.co_base = 0; ccl.co_split = 1 << 30; ccl.co_add = 0;
      ccl.h_out = h_hwc[l];
      conv_wmma<2, false><<<dim3(64, 1, B_), dim3(128), 0, stream>>>(ccl);
    }

    // w_last 1x1: x_gen (HWC fp32) + output frames  (1 wave/block, ct==0)
    ConvParams cg{};
    cg.taps = 1;
    cg.wpack = pk_wlast; cg.in = h_hwc[3]; cg.cpitch = NH_; cg.Cin = NH_;
    cg.in_bstride = hstride; cg.ci_off = 0;
    cg.co_base = 0; cg.co_split = 1 << 30; cg.co_add = 0;
    cg.xgen = xgen; cg.frames_out = out; cg.t = t; cg.gates = gates;
    conv_wmma<3, false><<<dim3(64, 1, B_), dim3(32), 0, stream>>>(cg);
  }

  // ---- loss ----
  loss_partial<<<dim3(1024), dim3(256), 0, stream>>>(out, frames, partial);
  loss_final<<<dim3(1), dim3(256), 0, stream>>>(partial, out + OUTN, 1024);
}
